// RITS_46076409151638
// MI455X (gfx1250) — matrix operand order repaired and checked by reading, not compiled, not run on hardware
//
#include <hip/hip_runtime.h>
#include <hip/hip_bf16.h>

// ---------------- problem dims ----------------
constexpr int RB = 512;   // batch
constexpr int RT = 256;   // time
constexpr int RF = 128;   // features
constexpr int RH = 256;   // hidden
constexpr int RG = 1024;  // 4H (LSTM gates)
constexpr int KG = 512;   // gates GEMM K = F (c_c) + F (m) + H (hdec)

typedef unsigned short u16;
typedef __attribute__((ext_vector_type(16))) __bf16 v16bf;
typedef __attribute__((ext_vector_type(8)))  float  v8f;
typedef __attribute__((ext_vector_type(4)))  unsigned int u32x4;
typedef __attribute__((ext_vector_type(8)))  int         i32x8;
typedef __attribute__((ext_vector_type(4)))  int         i32x4;

#ifndef __has_builtin
#define __has_builtin(x) 0
#endif
#if __has_builtin(__builtin_amdgcn_tensor_load_to_lds) && \
    __has_builtin(__builtin_amdgcn_s_wait_tensorcnt)
#define USE_TDM 1
#else
#define USE_TDM 0
#endif

// ---------------- helpers ----------------
static __device__ inline u16 f2bf(float x) {
  union { float f; unsigned u; } a; a.f = x;
  unsigned r = a.u + 0x7FFFu + ((a.u >> 16) & 1u);   // round-to-nearest-even
  return (u16)(r >> 16);
}
static __device__ inline float sigm(float x) { return 1.f / (1.f + __expf(-x)); }

union FragCvt { uint4 u[2]; v16bf v; };

// A-matrix 16x32 bf16 fragment (ISA 7.12.2): lane L -> row M=L%16, half hl=L/16;
// elems 0..7 = K[k0+hl*8 .. +7], elems 8..15 = K[k0+16+hl*8 .. +7]
static __device__ inline v16bf ld_a(const u16* base, int ldk, int k0, int lane) {
  int m = lane & 15, hl = lane >> 4;
  const u16* rp = base + m * ldk + k0 + hl * 8;
  FragCvt c;
  c.u[0] = *reinterpret_cast<const uint4*>(rp);
  c.u[1] = *reinterpret_cast<const uint4*>(rp + 16);
  return c.v;
}
// B-matrix 32x16 bf16 fragment: lane L -> col N=n0+L%16; elems 0..7 = K[k0+hl*8 .. +7], elems 8..15 = K[k0+16+hl*8 .. +7].
// Weights stored as W[N][K] (i.e. B[k][n] = W[n][k]), so per lane: two runs of 8 contiguous K.
static __device__ inline v16bf ld_b(const u16* W, int ldk, int n0, int k0, int lane) {
  int n = n0 + (lane & 15), hl = lane >> 4;
  const u16* rp = W + (size_t)n * ldk + k0 + hl * 8;
  FragCvt c;
  c.u[0] = *reinterpret_cast<const uint4*>(rp);
  c.u[1] = *reinterpret_cast<const uint4*>(rp + 16);
  return c.v;
}
static __device__ inline v8f wmma_bf(v16bf a, v16bf b, v8f c) {
  // (neg_a, A, neg_b, B, c_mod, C, reuse_a, reuse_b)
  return __builtin_amdgcn_wmma_f32_16x16x32_bf16(false, a, false, b, (short)0, c,
                                                 false, false);
}

#if USE_TDM
// Tensor Data Mover: DMA a rows x cols f32 tile (row stride in elements) into LDS.
// D# layout per CDNA5 ISA (groups 2/3 zero => 2-D tile).
// 6-arg builtin form: (g0, g1, g2, g3, g4, cpol)
static __device__ inline void tdm_load_2d_f32(const float* gsrc, void* ldst,
                                              int rows, int cols,
                                              int row_stride_elts) {
  unsigned lds_off = (unsigned)(size_t)ldst;               // LDS byte offset (addr[31:0])
  unsigned long long ga = (unsigned long long)(size_t)gsrc;
  u32x4 g0;
  g0[0] = 1u;                                              // count=1 (valid), user mode
  g0[1] = lds_off;                                         // lds_addr (bytes)
  g0[2] = (unsigned)(ga & 0xFFFFFFFFu);                    // global_addr[31:0]
  g0[3] = (unsigned)((ga >> 32) & 0x01FFFFFFu)             // global_addr[56:32]
          | (2u << 30);                                    // type = 2 ("image")
  i32x8 g1;
  g1[0] = (int)(2u << 16);                 // data_size=2 (4 bytes), no mask/flags
  g1[1] = (int)((unsigned)cols << 16);     // tensor_dim0[15:0]
  g1[2] = (int)((unsigned)rows << 16);     // tensor_dim1[15:0]
  g1[3] = (int)((unsigned)cols << 16);     // tile_dim0
  g1[4] = (int)(unsigned)rows;             // tile_dim1 (tile_dim2 = 0)
  g1[5] = row_stride_elts;                 // tensor_dim0_stride[31:0]
  g1[6] = 0;                               // stride hi / dim1_stride lo
  g1[7] = 0;
  i32x4 gz4 = {0, 0, 0, 0};
  i32x8 gz8 = {0, 0, 0, 0, 0, 0, 0, 0};
  __builtin_amdgcn_tensor_load_to_lds(g0, g1, gz4, gz4, gz8, 0);
}
#endif

// ---------------- weight prep: f32 -> bf16 WMMA-B images ----------------
__global__ void rits_prep(const float* __restrict__ Wdh, const float* __restrict__ Wh,
                          const float* __restrict__ Wf,  const float* __restrict__ Wc,
                          const float* __restrict__ Wih, const float* __restrict__ Whh,
                          const float* __restrict__ bih, const float* __restrict__ bhh,
                          u16* Wdh_bf, u16* Wh_bf, u16* Wf_bf, u16* Wc_bf, u16* Wg_bf,
                          float* bg) {
  int i0 = blockIdx.x * blockDim.x + threadIdx.x;
  int stride = gridDim.x * blockDim.x;
  for (int idx = i0; idx < RG * KG; idx += stride) {
    // gates: Wg[n][0:256] = Wih[n][:], Wg[n][256:512] = Whh[n][:]
    int n = idx >> 9, k = idx & (KG - 1);
    float v = (k < 2 * RF) ? Wih[n * (2 * RF) + k] : Whh[n * RH + (k - 2 * RF)];
    Wg_bf[idx] = f2bf(v);
    if (idx < RH * RF) Wdh_bf[idx] = f2bf(Wdh[idx]);           // [H][F]
    if (idx < RF * RH) Wh_bf[idx]  = f2bf(Wh[idx]);            // [F][H]
    if (idx < RF * RF) {                                       // [F][F], zero diag
      int r = idx >> 7, c = idx & (RF - 1);
      Wf_bf[idx] = f2bf((r == c) ? 0.f : Wf[idx]);
    }
    if (idx < RF * 2 * RF) Wc_bf[idx] = f2bf(Wc[idx]);         // [F][2F]
    if (idx < RG) bg[idx] = bih[idx] + bhh[idx];
  }
}

// ---------------- per-timestep mask denominator ----------------
__global__ void rits_maskden(const float* __restrict__ mask, float* den) {
  __shared__ float s[256];
  int t = blockIdx.x;
  float acc = 0.f;
  for (int i = threadIdx.x; i < RB * RF; i += 256) {
    int b = i >> 7, f = i & (RF - 1);
    acc += mask[((size_t)b * RT + t) * RF + f];
  }
  s[threadIdx.x] = acc;
  __syncthreads();
  for (int w = 128; w > 0; w >>= 1) {
    if (threadIdx.x < w) s[threadIdx.x] += s[threadIdx.x + w];
    __syncthreads();
  }
  if (threadIdx.x == 0) den[t] = s[0];
}

// ---------------- one RITS timestep: 32 blocks x 128 threads (4 waves) ----------------
__global__ __launch_bounds__(128) void rits_step(
    int t,
    const float* __restrict__ values, const float* __restrict__ mask,
    const float* __restrict__ deltas,
    const float* __restrict__ bdh, const float* __restrict__ Wdx,
    const float* __restrict__ bdx, const float* __restrict__ bh,
    const float* __restrict__ bf_, const float* __restrict__ bc,
    const u16* __restrict__ Wdh_bf, const u16* __restrict__ Wh_bf,
    const u16* __restrict__ Wf_bf,  const u16* __restrict__ Wc_bf,
    const u16* __restrict__ Wg_bf,  const float* __restrict__ bg,
    float* h_state, float* c_state, float* num_acc, float* out_imp) {
  __shared__ __align__(16) u16 A_d [16 * RF];        // deltas (A for gamma_h)
  __shared__ __align__(16) u16 A_gx[16 * 2 * RF];    // [gamma_x | m] (A for alpha)
  __shared__ __align__(16) u16 A_x [16 * RF];        // x_c (A for z_h)
  __shared__ __align__(16) u16 A_g [16 * KG];        // [c_c | m | hdec] (A for x_h & gates)
  __shared__ __align__(16) float xS[16 * RF];        // values tile (TDM target)
  __shared__ __align__(16) float mS[16 * RF];        // mask tile (TDM target)
  __shared__ float red[3 * 128];

  const int tid  = threadIdx.x;
  const int lane = tid & 31;
  const int wave = tid >> 5;
  const int b0   = blockIdx.x * 16;

  // ---- stage 0: async x/m tile DMA to LDS; stage deltas, gamma_x ----
#if USE_TDM
  if (wave == 0) {
    tdm_load_2d_f32(values + ((size_t)b0 * RT + t) * RF, xS, 16, RF, RT * RF);
    tdm_load_2d_f32(mask   + ((size_t)b0 * RT + t) * RF, mS, 16, RF, RT * RF);
  }
#endif
  {
    const int col = tid;  // 0..127
    const float wdxd = Wdx[col * RF + col];
    const float bdxc = bdx[col];
    for (int r = 0; r < 16; ++r) {
      size_t gi = ((size_t)(b0 + r) * RT + t) * RF + col;
#if !USE_TDM
      xS[r * RF + col] = values[gi];
      mS[r * RF + col] = mask[gi];
#endif
      float d = deltas[gi];
      A_d[r * RF + col] = f2bf(d);
      A_gx[r * 2 * RF + col] = f2bf(__expf(-fmaxf(0.f, d * wdxd + bdxc)));
    }
    if (t + 1 < RT) {  // warm next step (global_prefetch_b8)
      __builtin_prefetch(values + ((size_t)b0 * RT + (t + 1)) * RF + tid, 0, 0);
      __builtin_prefetch(deltas + ((size_t)b0 * RT + (t + 1)) * RF + tid, 0, 0);
    }
  }
#if USE_TDM
  if (wave == 0) __builtin_amdgcn_s_wait_tensorcnt(0);  // drain TDM before barrier
#endif
  __syncthreads();

  // ---- stage 0b: m sections of A matrices (consumed only after next barrier) ----
  {
    const int col = tid;
    for (int r = 0; r < 16; ++r) {
      u16 mb = f2bf(mS[r * RF + col]);
      A_gx[r * 2 * RF + RF + col] = mb;
      A_g [r * KG + RF + col]     = mb;
    }
  }

  // ---- stage 1: gamma_h = exp(-relu(d @ Wdh^T + bdh)); hdec = h*gamma_h -> A_g[:,256:] ----
  for (int jt = 0; jt < 4; ++jt) {
    int n0 = (wave * 4 + jt) * 16;  // over H
    v8f acc = {};
#pragma unroll
    for (int ks = 0; ks < RF / 32; ++ks)
      acc = wmma_bf(ld_a(A_d, RF, ks * 32, lane),
                    ld_b(Wdh_bf, RF, n0, ks * 32, lane), acc);
    int n = n0 + (lane & 15), hl = lane >> 4;
#pragma unroll
    for (int r = 0; r < 8; ++r) {
      int M = r + 8 * hl;
      float g  = __expf(-fmaxf(0.f, acc[r] + bdh[n]));
      float hd = h_state[(size_t)(b0 + M) * RH + n] * g;
      A_g[M * KG + 2 * RF + n] = f2bf(hd);
    }
  }
  __syncthreads();

  // ---- stage 2: x_h = hdec @ Wh^T + bh ; x_c = m*x + (1-m)*x_h ----
  float s1 = 0.f;
  float xhr[2][8];
#pragma unroll
  for (int jt = 0; jt < 2; ++jt) {
    int n0 = (wave * 2 + jt) * 16;  // over F
    v8f acc = {};
#pragma unroll
    for (int ks = 0; ks < RH / 32; ++ks)
      acc = wmma_bf(ld_a(A_g + 2 * RF, KG, ks * 32, lane),
                    ld_b(Wh_bf, RH, n0, ks * 32, lane), acc);
    int n = n0 + (lane & 15), hl = lane >> 4;
#pragma unroll
    for (int r = 0; r < 8; ++r) {
      int M = r + 8 * hl;
      float xh = acc[r] + bh[n];
      xhr[jt][r] = xh;
      float x = xS[M * RF + n], m = mS[M * RF + n];
      s1 += fabsf(xh - x) * m;
      A_x[M * RF + n] = f2bf(m * x + (1.f - m) * xh);
    }
  }
  __syncthreads();

  // ---- stage 3: alpha = [gx|m]@Wc^T+bc ; z_h = x_c@Wfm^T+bf ; c_h ; c_c -> A_g / output ----
  float s2 = 0.f, s3 = 0.f;
#pragma unroll
  for (int jt = 0; jt < 2; ++jt) {
    int n0 = (wave * 2 + jt) * 16;  // same tiles as stage 2 -> xhr matches
    v8f accA = {}, accZ = {};
#pragma unroll
    for (int ks = 0; ks < (2 * RF) / 32; ++ks)
      accA = wmma_bf(ld_a(A_gx, 2 * RF, ks * 32, lane),
                     ld_b(Wc_bf, 2 * RF, n0, ks * 32, lane), accA);
#pragma unroll
    for (int ks = 0; ks < RF / 32; ++ks)
      accZ = wmma_bf(ld_a(A_x, RF, ks * 32, lane),
                     ld_b(Wf_bf, RF, n0, ks * 32, lane), accZ);
    int n = n0 + (lane & 15), hl = lane >> 4;
#pragma unroll
    for (int r = 0; r < 8; ++r) {
      int M = r + 8 * hl;
      float alpha = accA[r] + bc[n];
      float z     = accZ[r] + bf_[n];
      float x = xS[M * RF + n], m = mS[M * RF + n];
      float ch = alpha * z + (1.f - alpha) * xhr[jt][r];
      s2 += fabsf(z - x) * m;
      s3 += fabsf(ch - x) * m;
      float cc = m * x + (1.f - m) * ch;
      A_g[M * KG + n] = f2bf(cc);
      out_imp[((size_t)(b0 + M) * RT + t) * RF + n] = cc;  // mask*x + (1-mask)*imp
    }
  }
  red[tid] = s1; red[128 + tid] = s2; red[256 + tid] = s3;
  __syncthreads();  // also publishes A_g c_c section for stage 4
  if (tid < 3) {
    float s = 0.f;
    for (int i = 0; i < 128; ++i) s += red[tid * 128 + i];
    atomicAdd(&num_acc[t * 3 + tid], s);
  }

  // ---- stage 4: gates = [c_c|m|hdec] @ [Wih|Whh]^T + bih+bhh ; LSTM update ----
  for (int q = 0; q < 4; ++q) {
    int nh0 = (wave * 4 + q) * 16;  // hidden tile over H
    v8f ai = {}, af = {}, ag = {}, ao = {};
#pragma unroll 4
    for (int ks = 0; ks < KG / 32; ++ks) {
      v16bf a = ld_a(A_g, KG, ks * 32, lane);
      ai = wmma_bf(a, ld_b(Wg_bf, KG, 0 * RH + nh0, ks * 32, lane), ai);
      af = wmma_bf(a, ld_b(Wg_bf, KG, 1 * RH + nh0, ks * 32, lane), af);
      ag = wmma_bf(a, ld_b(Wg_bf, KG, 2 * RH + nh0, ks * 32, lane), ag);
      ao = wmma_bf(a, ld_b(Wg_bf, KG, 3 * RH + nh0, ks * 32, lane), ao);
    }
    int n = nh0 + (lane & 15), hl = lane >> 4;
#pragma unroll
    for (int r = 0; r < 8; ++r) {
      int M = r + 8 * hl;
      size_t hi = (size_t)(b0 + M) * RH + n;
      float ig = sigm(ai[r] + bg[0 * RH + n]);
      float fg = sigm(af[r] + bg[1 * RH + n]);
      float gg = tanhf(ag[r] + bg[2 * RH + n]);
      float og = sigm(ao[r] + bg[3 * RH + n]);
      float cn = fg * c_state[hi] + ig * gg;
      c_state[hi] = cn;
      h_state[hi] = og * tanhf(cn);
    }
  }
}

// ---------------- loss finalize ----------------
// x_loss = sum_t(mae1+mae2) + sum_t (T-t)*mae3   (running mae_loss re-added each step)
__global__ void rits_finalize(const float* __restrict__ num_acc,
                              const float* __restrict__ den, float* out_tail) {
  __shared__ float sx[RT], sm[RT];
  int t = threadIdx.x;
  float d  = den[t] + 1e-9f;
  float m1 = num_acc[t * 3 + 0] / d;
  float m2 = num_acc[t * 3 + 1] / d;
  float m3 = num_acc[t * 3 + 2] / d;
  sx[t] = m1 + m2 + (float)(RT - t) * m3;
  sm[t] = m3;
  __syncthreads();
  for (int w = 128; w > 0; w >>= 1) {
    if (t < w) { sx[t] += sx[t + w]; sm[t] += sm[t + w]; }
    __syncthreads();
  }
  if (t == 0) {
    out_tail[0] = sx[0] / (RT * 3.0f);  // reconstruction_loss
    out_tail[1] = sm[0] / (float)RT;    // reconstruction_MAE
  }
}

// ---------------- launch ----------------
extern "C" void kernel_launch(void* const* d_in, const int* in_sizes, int n_in,
                              void* d_out, int out_size, void* d_ws, size_t ws_size,
                              hipStream_t stream) {
  const float* values = (const float*)d_in[0];
  const float* mask   = (const float*)d_in[1];
  const float* deltas = (const float*)d_in[2];
  const float* Wdh = (const float*)d_in[3];
  const float* bdh = (const float*)d_in[4];
  const float* Wdx = (const float*)d_in[5];
  const float* bdx = (const float*)d_in[6];
  const float* Wh  = (const float*)d_in[7];
  const float* bh  = (const float*)d_in[8];
  const float* Wf  = (const float*)d_in[9];
  const float* bf  = (const float*)d_in[10];
  const float* Wc  = (const float*)d_in[11];
  const float* bc  = (const float*)d_in[12];
  const float* Wih = (const float*)d_in[13];
  const float* Whh = (const float*)d_in[14];
  const float* bih = (const float*)d_in[15];
  const float* bhh = (const float*)d_in[16];
  (void)in_sizes; (void)n_in; (void)out_size; (void)ws_size;

  char* p = (char*)d_ws;
  float* h_state = (float*)p; p += (size_t)RB * RH * 4;
  float* c_state = (float*)p; p += (size_t)RB * RH * 4;
  float* num_acc = (float*)p; p += 3 * RT * 4;
  float* den     = (float*)p; p += RT * 4;
  u16* Wdh_bf = (u16*)p; p += (size_t)RH * RF * 2;
  u16* Wh_bf  = (u16*)p; p += (size_t)RF * RH * 2;
  u16* Wf_bf  = (u16*)p; p += (size_t)RF * RF * 2;
  u16* Wc_bf  = (u16*)p; p += (size_t)RF * 2 * RF * 2;
  u16* Wg_bf  = (u16*)p; p += (size_t)RG * KG * 2;
  float* bg   = (float*)p; p += (size_t)RG * 4;

  // zero h0, c0 and loss accumulators (deterministic per call; capture-safe)
  (void)hipMemsetAsync(d_ws, 0, ((size_t)RB * RH * 2 + 3 * RT) * 4, stream);

  rits_prep<<<512, 256, 0, stream>>>(Wdh, Wh, Wf, Wc, Wih, Whh, bih, bhh,
                                     Wdh_bf, Wh_bf, Wf_bf, Wc_bf, Wg_bf, bg);
  rits_maskden<<<RT, 256, 0, stream>>>(mask, den);

  float* out_imp = (float*)d_out;
  for (int t = 0; t < RT; ++t) {
    rits_step<<<RB / 16, 128, 0, stream>>>(
        t, values, mask, deltas, bdh, Wdx, bdx, bh, bf, bc,
        Wdh_bf, Wh_bf, Wf_bf, Wc_bf, Wg_bf, bg,
        h_state, c_state, num_acc, out_imp);
  }
  rits_finalize<<<1, RT, 0, stream>>>(num_acc, den,
                                      out_imp + (size_t)RB * RT * RF);
}
